// custom_MHSA_82549271429605
// MI455X (gfx1250) — matrix operand order repaired and checked by reading, not compiled, not run on hardware
//
#include <hip/hip_runtime.h>
#include <hip/hip_bf16.h>

#define BB 4
#define NN 2048
#define DMODEL 768
#define NH 12
#define HD 64
#define MROWS (BB*NN)   // 8192

typedef __attribute__((ext_vector_type(16))) __bf16       v16bf;
typedef __attribute__((ext_vector_type(8)))  float        v8f;
typedef __attribute__((ext_vector_type(4)))  float        v4f;
typedef __attribute__((ext_vector_type(4)))  unsigned int v4u;
typedef __attribute__((ext_vector_type(8)))  int          v8i;
typedef __attribute__((ext_vector_type(4)))  int          v4i;

#if __has_builtin(__builtin_amdgcn_tensor_load_to_lds) && \
    __has_builtin(__builtin_amdgcn_s_wait_tensorcnt)
#define USE_TDM 1
#else
#define USE_TDM 0
#endif

union V16 {
  v16bf          v;
  unsigned short u[16];
  unsigned int   d[8];
  v4u            q4[2];
};

static __device__ __forceinline__ unsigned short f2bf(float f) {
  unsigned int u = __float_as_uint(f);
  u += 0x7FFFu + ((u >> 16) & 1u);   // round-to-nearest-even
  return (unsigned short)(u >> 16);
}

#if USE_TDM
// ---------------------------------------------------------------------------
// Tensor Data Mover: 2-D tile load global->LDS.  D# packed per CDNA5 ISA
// §8.3-8.6.  All sizes in 4-byte units (data_size=4B).  Groups 2/3 zero
// (tile_dim2/3 = 0 -> 2-D tile).  Issued by one wave; tracked by TENSORcnt.
// 6-arg builtin form (clang-23 / therock-10.0 toolchain).
// ---------------------------------------------------------------------------
static __device__ __forceinline__ void tdm_load_2d(
    const void* lds_ptr, const void* gptr,
    unsigned int tdim0, unsigned int tdim1,   // tensor size (units, rows)
    unsigned int tile0, unsigned int tile1,   // tile size   (units, rows)
    unsigned int stride0)                     // row pitch   (units)
{
  const unsigned int      lds_off = (unsigned int)(uintptr_t)lds_ptr;
  const unsigned long long ga     = (unsigned long long)(uintptr_t)gptr;
  v4u g0;
  g0.x = 1u;                                          // count=1 (valid), no gather
  g0.y = lds_off;                                     // D#.lds_addr (bytes)
  g0.z = (unsigned int)ga;                            // global_addr[31:0]
  g0.w = (unsigned int)((ga >> 32) & 0x1FFFFFFull)    // global_addr[56:32]
       | (2u << 30);                                  // type=2 ("image")
  v8i g1;
  g1[0] = (int)(2u << 16);                            // wg_mask=0, data_size=4B
  g1[1] = (int)((tdim0 & 0xFFFFu) << 16);             // tensor_dim0[15:0]
  g1[2] = (int)((tdim0 >> 16) | ((tdim1 & 0xFFFFu) << 16));
  g1[3] = (int)((tdim1 >> 16) | (tile0 << 16));       // tensor_dim1[31:16] | tile_dim0
  g1[4] = (int)tile1;                                 // tile_dim1, tile_dim2=0
  g1[5] = (int)stride0;                               // tensor_dim0_stride[31:0]
  g1[6] = 0;
  g1[7] = 0;
  const v4i z4 = {0, 0, 0, 0};
  const v8i z8 = {0, 0, 0, 0, 0, 0, 0, 0};
  __builtin_amdgcn_tensor_load_to_lds(g0, g1, z4, z4, z8, 0);
}
#endif

// ---------------------------------------------------------------------------
// Kernel 0: one-shot f32 -> bf16 conversion (hoists ALL cvt VALU out of GEMMs)
// ---------------------------------------------------------------------------
__global__ __launch_bounds__(256) void mhsa_cvt_bf16(
    const float* __restrict__ src, unsigned short* __restrict__ dst, int n)
{
  int i = (blockIdx.x * 256 + threadIdx.x) * 8;
  if (i >= n) return;
  v4f a = *(const v4f*)(src + i);
  v4f b = *(const v4f*)(src + i + 4);
  v4u o;
  o.x = (unsigned int)f2bf(a[0]) | ((unsigned int)f2bf(a[1]) << 16);
  o.y = (unsigned int)f2bf(a[2]) | ((unsigned int)f2bf(a[3]) << 16);
  o.z = (unsigned int)f2bf(b[0]) | ((unsigned int)f2bf(b[1]) << 16);
  o.w = (unsigned int)f2bf(b[2]) | ((unsigned int)f2bf(b[3]) << 16);
  *(v4u*)(dst + i) = o;
}

// ---------------------------------------------------------------------------
// Kernel 1: QKV projection GEMM, all-bf16 operands.
// Block = 256 thr (8 waves). Each wave: 32(M) x 64(O) via 8 WMMA accumulators.
// W tile (32K x 64O, 4KB) staged in LDS, TDM double-buffered when available.
// q,k stored [B,H,N,64]; v stored transposed [B,H,64,N].
// ---------------------------------------------------------------------------
__global__ __launch_bounds__(256) void mhsa_qkv_gemm(
    const unsigned short* __restrict__ xbf,
    const unsigned short* __restrict__ wbf,   // [3][768][768]
    const float* __restrict__ bq, const float* __restrict__ bk,
    const float* __restrict__ bv,
    unsigned short* __restrict__ qws, unsigned short* __restrict__ kws,
    unsigned short* __restrict__ vws)
{
#if USE_TDM
  __shared__ __align__(16) unsigned short sW[2][64 * 32];  // double buffer, 8KB
#else
  __shared__ __align__(16) unsigned short sW[1][64 * 32];
#endif

  const int mat = blockIdx.z;
  const unsigned short* wmat = wbf + (size_t)mat * DMODEL * DMODEL;
  const float* bb = (mat == 0) ? bq : (mat == 1) ? bk : bv;
  unsigned short* outp = (mat == 0) ? qws : (mat == 1) ? kws : vws;

  const int t    = threadIdx.x;
  const int wave = t >> 5;
  const int lane = t & 31;
  const int hh   = lane >> 4;
  const int l15  = lane & 15;

  const int m0 = blockIdx.x * 256 + wave * 32;   // 2 row tiles per wave
  const int o0 = blockIdx.y * 64;

  v8f acc[2][4] = {};

#if USE_TDM
  // prologue: DMA first W tile (64 rows x 64B, pitch 1536B) into buffer 0
  if (wave == 0) {
    tdm_load_2d(&sW[0][0], wmat + (size_t)o0 * DMODEL,
                /*tdim0=*/384, /*tdim1=*/DMODEL, /*tile0=*/16, /*tile1=*/64,
                /*stride0=*/384);
    __builtin_amdgcn_s_wait_tensorcnt(0);
  }
  __syncthreads();
#else
  const int co = t >> 2, ck = (t & 3) * 8;       // cooperative chunk ownership
#endif

  for (int k0 = 0; k0 < DMODEL; k0 += 32) {
#if USE_TDM
    const int cur = (k0 >> 5) & 1, nxt = cur ^ 1;
    if (wave == 0 && k0 + 32 < DMODEL)           // DMA next tile into other buf
      tdm_load_2d(&sW[nxt][0], wmat + (size_t)o0 * DMODEL + (k0 + 32),
                  384, DMODEL, 16, 64, 384);
    const unsigned short* sWb = &sW[cur][0];
#else
    v4u wchunk = *(const v4u*)(wmat + (size_t)(o0 + co) * DMODEL + k0 + ck);
    if (k0 + 32 < DMODEL)
      __builtin_prefetch(wmat + (size_t)(o0 + co) * DMODEL + k0 + 32 + ck, 0, 1);
    __syncthreads();
    *(v4u*)(&sW[0][0] + co * 32 + ck) = wchunk;
    __syncthreads();
    const unsigned short* sWb = &sW[0][0];
#endif

    // A tiles (x rows), bf16 A-layout: two contiguous 16B chunks per lane
    V16 a[2];
#pragma unroll
    for (int r2 = 0; r2 < 2; ++r2) {
      const unsigned short* p = xbf + (size_t)(m0 + 16 * r2 + l15) * DMODEL + k0 + hh * 8;
      a[r2].q4[0] = *(const v4u*)p;        // K = 8h + 0..7
      a[r2].q4[1] = *(const v4u*)(p + 16); // K = 16 + 8h + 0..7
    }

#pragma unroll
    for (int j = 0; j < 4; ++j) {
      V16 bm;   // B layout: lane col o = 16j + l15, elems kk = 8h + 0..7, 16 + 8h + 0..7
      const unsigned short* p = sWb + (16 * j + l15) * 32 + hh * 8;
      bm.q4[0] = *(const v4u*)p;
      bm.q4[1] = *(const v4u*)(p + 16);
#pragma unroll
      for (int r2 = 0; r2 < 2; ++r2)
        acc[r2][j] = __builtin_amdgcn_wmma_f32_16x16x32_bf16(
            false, a[r2].v, false, bm.v, (short)0, acc[r2][j], false, false);
    }

#if USE_TDM
    if (wave == 0 && k0 + 32 < DMODEL)
      __builtin_amdgcn_s_wait_tensorcnt(0);    // next buffer landed
    __syncthreads();                           // all waves done with cur buf
#endif
  }

  // bias + head-split store (C layout: VGPR r -> M = r + 8h, N = l15)
#pragma unroll
  for (int r2 = 0; r2 < 2; ++r2) {
#pragma unroll
    for (int j = 0; j < 4; ++j) {
      const int o = o0 + j * 16 + l15;
      const float bias = bb[o];
      const int head = o >> 6, d = o & 63;
#pragma unroll
      for (int r = 0; r < 8; ++r) {
        const int m = m0 + 16 * r2 + r + 8 * hh;
        const int b = m >> 11, n = m & (NN - 1);
        const unsigned short bf = f2bf(acc[r2][j][r] + bias);
        size_t idx;
        if (mat == 2) idx = (((size_t)(b * NH + head) * HD + d) * NN + n);
        else          idx = (((size_t)(b * NH + head) * NN + n) * HD + d);
        outp[idx] = bf;
      }
    }
  }
}

// ---------------------------------------------------------------------------
// Kernel 2: flash attention, transposed-S trick, K/V^T tiles TDM-staged in LDS
// (double-buffered).  Block = 8 waves, same (b,h); wave owns 16 queries.
//   S^T(32x16) = K x Q^T   (4 WMMA)   -> lane-local online softmax
//   O^T(64x16) += V^T x P^T (4 WMMA)
// ---------------------------------------------------------------------------
__global__ __launch_bounds__(256) void mhsa_flash_attn(
    const unsigned short* __restrict__ qws,
    const unsigned short* __restrict__ kws,
    const unsigned short* __restrict__ vws,   // [B,H,64,N]
    float* __restrict__ out)
{
#if USE_TDM
  __shared__ __align__(16) unsigned short sK[2][32 * 64];  // [key][d]
  __shared__ __align__(16) unsigned short sV[2][64 * 32];  // [d][key]
#else
  __shared__ __align__(16) unsigned short sK[1][32 * 64];
  __shared__ __align__(16) unsigned short sV[1][64 * 32];
#endif

  const int t    = threadIdx.x;
  const int wave = t >> 5;
  const int lane = t & 31;
  const int hh   = lane >> 4;
  const int l15  = lane & 15;

  const int bhid = blockIdx.y;
  const int q0   = (blockIdx.x * 8 + wave) * 16;

  const unsigned short* Qp = qws + (size_t)bhid * NN * HD;
  const unsigned short* Kp = kws + (size_t)bhid * NN * HD;
  const unsigned short* Vp = vws + (size_t)bhid * HD * NN;

#if !USE_TDM
  const int kKey = t >> 3, kDc = (t & 7) * 8;   // K tile: 32 rows x 8 chunks
  const int vD   = t >> 2, vKc = (t & 3) * 8;   // V tile: 64 rows x 4 chunks
#endif

  // Q^T B-operand (lane col q = q0+l15, elems d = 32ks + A-pattern), in regs
  V16 qb[2];
#pragma unroll
  for (int ks = 0; ks < 2; ++ks) {
    const unsigned short* p = Qp + (size_t)(q0 + l15) * HD + ks * 32 + hh * 8;
    qb[ks].q4[0] = *(const v4u*)p;
    qb[ks].q4[1] = *(const v4u*)(p + 16);
  }

  v8f   oacc[4] = {};
  float mrun = -__builtin_inff();
  float lrun = 0.0f;
  const float cfac = 0.18033688011112042f;      // (1/sqrt(64)) * log2(e)

#if USE_TDM
  // prologue: DMA K tile (32 x 128B, pitch 128B) + V^T tile (64 x 64B, pitch 4KB)
  if (wave == 0) {
    tdm_load_2d(&sK[0][0], Kp, /*tdim0=*/32,   /*tdim1=*/NN, 32, 32, 32);
    tdm_load_2d(&sV[0][0], Vp, /*tdim0=*/1024, /*tdim1=*/64, 16, 64, 1024);
    __builtin_amdgcn_s_wait_tensorcnt(0);
  }
  __syncthreads();
#endif

  for (int k0 = 0; k0 < NN; k0 += 32) {
#if USE_TDM
    const int cur = (k0 >> 5) & 1, nxt = cur ^ 1;
    if (wave == 0 && k0 + 32 < NN) {
      tdm_load_2d(&sK[nxt][0], Kp + (size_t)(k0 + 32) * HD, 32, NN, 32, 32, 32);
      tdm_load_2d(&sV[nxt][0], Vp + (k0 + 32),          1024, 64, 16, 64, 1024);
    }
    const unsigned short* sKb = &sK[cur][0];
    const unsigned short* sVb = &sV[cur][0];
#else
    v4u kchunk = *(const v4u*)(Kp + (size_t)(k0 + kKey) * HD + kDc);
    v4u vchunk = *(const v4u*)(Vp + (size_t)vD * NN + k0 + vKc);
    if (k0 + 32 < NN) {
      __builtin_prefetch(Kp + (size_t)(k0 + 32 + kKey) * HD + kDc, 0, 1);
      __builtin_prefetch(Vp + (size_t)vD * NN + k0 + 32 + vKc, 0, 1);
    }
    __syncthreads();
    *(v4u*)(&sK[0][0] + kKey * 64 + kDc) = kchunk;
    *(v4u*)(&sV[0][0] + vD * 32 + vKc)   = vchunk;
    __syncthreads();
    const unsigned short* sKb = &sK[0][0];
    const unsigned short* sVb = &sV[0][0];
#endif

    // ---- S^T: two 16x16 key tiles from LDS
    v8f s[2];
#pragma unroll
    for (int tt = 0; tt < 2; ++tt) {
      v8f c = {};
#pragma unroll
      for (int ks = 0; ks < 2; ++ks) {
        V16 a;   // lane row key = 16tt + l15; elems d = 32ks + A-pattern
        const unsigned short* p = sKb + (16 * tt + l15) * 64 + ks * 32 + hh * 8;
        a.q4[0] = *(const v4u*)p;
        a.q4[1] = *(const v4u*)(p + 16);
        c = __builtin_amdgcn_wmma_f32_16x16x32_bf16(
            false, a.v, false, qb[ks].v, (short)0, c, false, false);
      }
      s[tt] = c;
    }

    // ---- online softmax in exp2 domain; lane holds fixed query col
    float sl[16];
#pragma unroll
    for (int r = 0; r < 8; ++r) { sl[r] = s[0][r] * cfac; sl[8 + r] = s[1][r] * cfac; }
    float bm = sl[0];
#pragma unroll
    for (int i = 1; i < 16; ++i) bm = fmaxf(bm, sl[i]);
    bm = fmaxf(bm, __shfl_xor(bm, 16, 32));
    const float mnew  = fmaxf(mrun, bm);
    const float alpha = __builtin_amdgcn_exp2f(mrun - mnew);
    mrun = mnew;

    float p[16]; float psum = 0.0f;
#pragma unroll
    for (int i = 0; i < 16; ++i) {
      p[i] = __builtin_amdgcn_exp2f(sl[i] - mnew);
      psum += p[i];
    }
    lrun = lrun * alpha + psum;
#pragma unroll
    for (int j = 0; j < 4; ++j)
#pragma unroll
      for (int r = 0; r < 8; ++r) oacc[j][r] *= alpha;

    // ---- pack P^T into B layout (lane needs keys 16h + 0..15 of block)
    unsigned int pa[4], pb[4], ra[4], rb[4];
#pragma unroll
    for (int i = 0; i < 4; ++i) {
      pa[i] = (unsigned int)f2bf(p[2 * i])     | ((unsigned int)f2bf(p[2 * i + 1]) << 16);
      pb[i] = (unsigned int)f2bf(p[8 + 2 * i]) | ((unsigned int)f2bf(p[9 + 2 * i]) << 16);
    }
#pragma unroll
    for (int i = 0; i < 4; ++i) {
      ra[i] = (unsigned int)__shfl_xor((int)pa[i], 16, 32);
      rb[i] = (unsigned int)__shfl_xor((int)pb[i], 16, 32);
    }
    V16 pt;
#pragma unroll
    for (int i = 0; i < 4; ++i) {
      pt.d[i]     = hh ? rb[i] : pa[i];
      pt.d[4 + i] = hh ? pb[i] : ra[i];
    }

    // ---- O^T += V^T x P^T   (4 d-tiles, A from LDS)
#pragma unroll
    for (int j = 0; j < 4; ++j) {
      V16 a;   // lane row d = 16j + l15; elems keys = 16h + 0..15
      const unsigned short* p2 = sVb + (16 * j + l15) * 32 + hh * 16;
      a.q4[0] = *(const v4u*)p2;
      a.q4[1] = *(const v4u*)(p2 + 8);
      oacc[j] = __builtin_amdgcn_wmma_f32_16x16x32_bf16(
          false, a.v, false, pt.v, (short)0, oacc[j], false, false);
    }

#if USE_TDM
    if (wave == 0 && k0 + 32 < NN)
      __builtin_amdgcn_s_wait_tensorcnt(0);    // next buffers landed
    __syncthreads();                           // all waves done with cur bufs
#endif
  }

  // ---- normalize and store out[b, q, head*64 + d]
  const float ltot = lrun + __shfl_xor(lrun, 16, 32);
  const float inv  = 1.0f / ltot;
  const int b = bhid / NH, head = bhid % NH;
  const int q = q0 + l15;
  float* orow = out + ((size_t)b * NN + q) * DMODEL + head * HD;
#pragma unroll
  for (int j = 0; j < 4; ++j) {
    float* dp = orow + 16 * j + 8 * hh;    // d = 16j + 8h + r, r contiguous
    v4f lo = { oacc[j][0] * inv, oacc[j][1] * inv, oacc[j][2] * inv, oacc[j][3] * inv };
    v4f hi = { oacc[j][4] * inv, oacc[j][5] * inv, oacc[j][6] * inv, oacc[j][7] * inv };
    *(v4f*)dp       = lo;
    *(v4f*)(dp + 4) = hi;
  }
}

// ---------------------------------------------------------------------------
extern "C" void kernel_launch(void* const* d_in, const int* in_sizes, int n_in,
                              void* d_out, int out_size, void* d_ws, size_t ws_size,
                              hipStream_t stream) {
  (void)in_sizes; (void)n_in; (void)out_size; (void)ws_size;
  const float* x  = (const float*)d_in[0];
  const float* wq = (const float*)d_in[1];
  const float* bq = (const float*)d_in[2];
  const float* wk = (const float*)d_in[3];
  const float* bk = (const float*)d_in[4];
  const float* wv = (const float*)d_in[5];
  const float* bv = (const float*)d_in[6];
  float* out = (float*)d_out;

  // workspace layout (bf16 elements):
  //   xbf [8192*768] | wbf [3*768*768] | q | k | v  ([B,H,N,64]/[B,H,64,N])
  const size_t X_ELEMS   = (size_t)MROWS * DMODEL;        // 6,291,456
  const size_t W_ELEMS   = (size_t)DMODEL * DMODEL;       //   589,824
  const size_t QKV_ELEMS = (size_t)BB * NH * NN * HD;     // 6,291,456
  unsigned short* xbf = (unsigned short*)d_ws;
  unsigned short* wbf = xbf + X_ELEMS;
  unsigned short* qws = wbf + 3 * W_ELEMS;
  unsigned short* kws = qws + QKV_ELEMS;
  unsigned short* vws = kws + QKV_ELEMS;

  // one-shot conversions (bandwidth-trivial: ~33 MB total)
  mhsa_cvt_bf16<<<dim3(X_ELEMS / 2048), 256, 0, stream>>>(x, xbf, (int)X_ELEMS);
  mhsa_cvt_bf16<<<dim3(W_ELEMS / 2048), 256, 0, stream>>>(wq, wbf,               (int)W_ELEMS);
  mhsa_cvt_bf16<<<dim3(W_ELEMS / 2048), 256, 0, stream>>>(wk, wbf + W_ELEMS,     (int)W_ELEMS);
  mhsa_cvt_bf16<<<dim3(W_ELEMS / 2048), 256, 0, stream>>>(wv, wbf + 2 * W_ELEMS, (int)W_ELEMS);

  // QKV projection: 32 row-blocks x 12 col-blocks x {q,k,v}
  mhsa_qkv_gemm<<<dim3(MROWS / 256, DMODEL / 64, 3), 256, 0, stream>>>(
      xbf, wbf, bq, bk, bv, qws, kws, vws);

  // Flash attention: 16 q-blocks x (B*H)
  mhsa_flash_attn<<<dim3(NN / 16 / 8, BB * NH), 256, 0, stream>>>(
      qws, kws, vws, out);
}
